// Transformer_4647154614454
// MI455X (gfx1250) — matrix operand order repaired and checked by reading, not compiled, not run on hardware
//
#include <hip/hip_runtime.h>

// ---------------------------------------------------------------------------
// Transformer encoder-decoder forward for MI455X (gfx1250), wave32 + WMMA.
// All GEMMs (projections, FFN, attention QK^T and PV) use
// v_wmma_f32_16x16x32_f16 with f32 accumulation. Attention is flash-style
// (no materialized [S,S] scores). V is produced transposed (V^T [E, B*S]) by
// its projection GEMM so attention B-fragments are contiguous 32B loads.
// ---------------------------------------------------------------------------

typedef _Float16 h16;
typedef __attribute__((ext_vector_type(16))) _Float16 v16h;
typedef __attribute__((ext_vector_type(8)))  float    v8f;

__device__ __forceinline__ v8f wmma_f16f32(v16h a, v16h b, v8f c) {
  // (neg_a, A, neg_b, B, c_mod, C, reuse_a, reuse_b)
  return __builtin_amdgcn_wmma_f32_16x16x32_f16(false, a, false, b, (short)0, c,
                                                false, false);
}

union Frag { uint4 u[2]; v16h v; };
union H8   { h16 hh[8]; uint4 u; };

// ---------------- elementwise f32 -> f16 ----------------
__global__ void k_cvt_h(const float* __restrict__ in, h16* __restrict__ out, int n) {
  int i = blockIdx.x * blockDim.x + threadIdx.x;
  if (i < n) out[i] = (h16)in[i];
}

// ---------------- transpose + convert: in f32 [K,N] -> out f16 [N,K] -------
__global__ void k_transpose_h(const float* __restrict__ in, h16* __restrict__ out,
                              int K, int N) {
  __shared__ float t[32][33];
  int n0 = blockIdx.x * 32, k0 = blockIdx.y * 32;
  int tx = threadIdx.x, ty = threadIdx.y;  // block (32,8)
#pragma unroll
  for (int i = 0; i < 32; i += 8)
    t[ty + i][tx] = in[(size_t)(k0 + ty + i) * N + (n0 + tx)];
  __syncthreads();
#pragma unroll
  for (int i = 0; i < 32; i += 8)
    out[(size_t)(n0 + ty + i) * K + (k0 + tx)] = (h16)t[tx][ty + i];
}

// ---------------- LayerNorm over last dim E --------------------------------
__global__ void k_ln(const float* __restrict__ in, const float* __restrict__ g,
                     const float* __restrict__ b, float* __restrict__ outF,
                     h16* __restrict__ outH, int E) {
  int row = blockIdx.x;
  int tid = threadIdx.x;  // 128 threads
  const float* x = in + (size_t)row * E;
  float s = 0.f, sq = 0.f;
  for (int e = tid; e < E; e += 128) { float v = x[e]; s += v; sq += v * v; }
  __shared__ float rs[128], rq[128];
  rs[tid] = s; rq[tid] = sq;
  __syncthreads();
  for (int o = 64; o > 0; o >>= 1) {
    if (tid < o) { rs[tid] += rs[tid + o]; rq[tid] += rq[tid + o]; }
    __syncthreads();
  }
  float mean = rs[0] / (float)E;
  float var  = rq[0] / (float)E - mean * mean;
  float rstd = rsqrtf(var + 1e-5f);
  for (int e = tid; e < E; e += 128) {
    float v = (x[e] - mean) * rstd * g[e] + b[e];
    if (outF) outF[(size_t)row * E + e] = v;
    if (outH) outH[(size_t)row * E + e] = (h16)v;
  }
}

// ---------------- sinusoidal positional embedding (in-place) ---------------
__global__ void k_pos(float* __restrict__ y, int S, int E, float invSqrtE, int total) {
  int idx = blockIdx.x * blockDim.x + threadIdx.x;
  if (idx >= total) return;
  int e = idx % E;
  int s = (idx / E) % S;
  float j  = (float)(e & ~1);  // pairs share the same frequency
  float wv = (float)s * __expf(-j * (9.210340371976184f / (float)E)); // ln(10000)
  float pe = (e & 1) ? __cosf(wv) : __sinf(wv);
  y[idx] = pe + y[idx] * invSqrtE;
}

// ---------------- WMMA GEMM: C[M,N] = A[M,K](f16) @ Wt[N,K]^T (f16) --------
// epilogue: + bias, + residual, relu; writes f32, f16, and/or f16-transposed
// (outHT is [N, M] row-major -- used for V so attention can load V^T directly)
__global__ void __launch_bounds__(256)
k_gemm(const h16* __restrict__ A, const h16* __restrict__ Wt,
       const float* __restrict__ bias, const float* __restrict__ res,
       float* __restrict__ outF, h16* __restrict__ outH, h16* __restrict__ outHT,
       int M, int N, int K, int relu) {
  int lane = threadIdx.x & 31;
  int wv   = blockIdx.x * 8 + (threadIdx.x >> 5);
  int tilesM = M >> 4, tilesN = N >> 6;
  if (wv >= tilesM * tilesN) return;
  int tm = wv % tilesM, tn = wv / tilesM;
  int m0 = tm << 4, n0 = tn << 6;
  int half = lane >> 4, l16 = lane & 15;
  int koff = half * 16;

  const h16* arow = A + (size_t)(m0 + l16) * K;
  v8f zero = {0.f, 0.f, 0.f, 0.f, 0.f, 0.f, 0.f, 0.f};
  v8f acc[4];
#pragma unroll
  for (int t = 0; t < 4; ++t) acc[t] = zero;

  for (int k0 = 0; k0 < K; k0 += 32) {
    Frag af;
    af.u[0] = *(const uint4*)(arow + k0 + koff);
    af.u[1] = *(const uint4*)(arow + k0 + 8 + koff);
#pragma unroll
    for (int t = 0; t < 4; ++t) {
      const h16* brow = Wt + (size_t)(n0 + t * 16 + l16) * K + k0 + half * 16;
      v16h bf = *(const v16h*)brow;  // 16 contiguous halves = 32B
      acc[t] = wmma_f16f32(af.v, bf, acc[t]);
    }
  }

#pragma unroll
  for (int t = 0; t < 4; ++t) {
    int n = n0 + t * 16 + l16;
    float bval = bias ? bias[n] : 0.f;
    H8 tr;
#pragma unroll
    for (int j = 0; j < 8; ++j) {
      int m = m0 + half * 8 + j;
      float v = acc[t][j] + bval;
      if (res)  v += res[(size_t)m * N + n];
      if (relu) v = v > 0.f ? v : 0.f;
      if (outF) outF[(size_t)m * N + n] = v;
      if (outH) outH[(size_t)m * N + n] = (h16)v;
      tr.hh[j] = (h16)v;
    }
    if (outHT)  // 8 consecutive m at fixed n -> one 16B store
      *(uint4*)(outHT + (size_t)n * M + m0 + half * 8) = tr.u;
  }
}

// ---------------- flash attention: one wave per (b, h, 16-query tile) ------
// Q,K: f16 [B*S, E]; Vt: f16 [E, B*S] (transposed); O: f16 [B*S, E].
// Head h occupies rows/cols h*64..h*64+63 of the E dim.
__global__ void __launch_bounds__(32)
k_attn(const h16* __restrict__ Q, const h16* __restrict__ Km,
       const h16* __restrict__ Vt, h16* __restrict__ O,
       int S, int E, int Mtot, float scale) {
  int lane = threadIdx.x;
  int qt = blockIdx.x, h = blockIdx.y, b = blockIdx.z;
  int ho = h * 64;
  int qbase = b * S + qt * 16;
  int kvb   = b * S;
  int half = lane >> 4, l16 = lane & 15, koff = half * 16;

  __shared__ h16 pt[16][40];  // P tile [16 q x 32 keys], padded

  // Q fragments (A layout), hd split 0..31 / 32..63
  const h16* qrow = Q + (size_t)(qbase + l16) * E + ho;
  Frag aq0, aq1;
  aq0.u[0] = *(const uint4*)(qrow + koff);
  aq0.u[1] = *(const uint4*)(qrow + 8 + koff);
  aq1.u[0] = *(const uint4*)(qrow + 32 + koff);
  aq1.u[1] = *(const uint4*)(qrow + 40 + koff);

  v8f zero = {0.f, 0.f, 0.f, 0.f, 0.f, 0.f, 0.f, 0.f};
  v8f acc[4];
#pragma unroll
  for (int t = 0; t < 4; ++t) acc[t] = zero;
  float rmax[8], rsum[8];
#pragma unroll
  for (int j = 0; j < 8; ++j) { rmax[j] = -1e30f; rsum[j] = 0.f; }

  for (int kb = 0; kb < S; kb += 32) {
    // S = Q K^T for two 16-key subtiles (K-dim = 64 -> 2 chained WMMAs each)
    v8f st[2];
#pragma unroll
    for (int sub = 0; sub < 2; ++sub) {
      const h16* krow = Km + (size_t)(kvb + kb + sub * 16 + l16) * E + ho + half * 16;
      v16h bk0 = *(const v16h*)krow;         // hd 0..31
      v16h bk1 = *(const v16h*)(krow + 32);  // hd 32..63
      v8f sz = zero;
      sz = wmma_f16f32(aq0.v, bk0, sz);
      sz = wmma_f16f32(aq1.v, bk1, sz);
      st[sub] = sz;
    }
    // flash softmax update per row (row r_j = half*8 + j, col = l16)
    float corr[8];
#pragma unroll
    for (int j = 0; j < 8; ++j) {
      float v0 = st[0][j] * scale, v1 = st[1][j] * scale;
      float mloc = fmaxf(v0, v1);
#pragma unroll
      for (int msk = 8; msk >= 1; msk >>= 1)
        mloc = fmaxf(mloc, __shfl_xor(mloc, msk, 32));
      float nm = fmaxf(rmax[j], mloc);
      float c  = __expf(rmax[j] - nm);
      float p0 = __expf(v0 - nm), p1 = __expf(v1 - nm);
      float ls = p0 + p1;
#pragma unroll
      for (int msk = 8; msk >= 1; msk >>= 1)
        ls += __shfl_xor(ls, msk, 32);
      rsum[j] = rsum[j] * c + ls;
      rmax[j] = nm;
      corr[j] = c;
      int r = half * 8 + j;
      pt[r][l16]      = (h16)p0;
      pt[r][16 + l16] = (h16)p1;
    }
#pragma unroll
    for (int t = 0; t < 4; ++t)
#pragma unroll
      for (int j = 0; j < 8; ++j) acc[t][j] *= corr[j];
    __syncthreads();  // pt written before A-fragment gather
    // P as A fragment [16 x 32]
    Frag ap;
    ap.u[0] = *(const uint4*)&pt[l16][koff];
    ap.u[1] = *(const uint4*)&pt[l16][8 + koff];
    // O += P @ V: B-fragment = 16 contiguous halves of V^T (32B load)
#pragma unroll
    for (int t = 0; t < 4; ++t) {
      const h16* vcol = Vt + (size_t)(ho + t * 16 + l16) * Mtot
                           + kvb + kb + half * 16;
      v16h bv = *(const v16h*)vcol;
      acc[t] = wmma_f16f32(ap.v, bv, acc[t]);
    }
    __syncthreads();  // pt reads done before next iteration overwrites it
  }
  // finalize: divide by row sums, store f16
  float inv[8];
#pragma unroll
  for (int j = 0; j < 8; ++j) inv[j] = 1.f / rsum[j];
#pragma unroll
  for (int t = 0; t < 4; ++t)
#pragma unroll
    for (int j = 0; j < 8; ++j) {
      int m = qbase + half * 8 + j;
      int n = ho + t * 16 + l16;
      O[(size_t)m * E + n] = (h16)(acc[t][j] * inv[j]);
    }
}

// ---------------------------------------------------------------------------
// Host orchestration
// ---------------------------------------------------------------------------
struct LP {
  const float *ln1_g, *ln1_b, *wq, *bq, *wk, *bk, *wv, *bv, *wo, *bo;
  const float *ln2_g, *ln2_b, *w1, *b1, *w2, *b2;
};

extern "C" void kernel_launch(void* const* d_in, const int* in_sizes, int n_in,
                              void* d_out, int out_size, void* d_ws, size_t ws_size,
                              hipStream_t stream) {
  constexpr int B = 2, S = 2048, DI = 512, E = 512, F = 2048, NH = 8, NL = 4;
  constexpr int M = B * S;  // 4096
  const float kInvSqrtE = 0.044194173824159216f;  // 1/sqrt(512)

  // -------- input pointers (top-level dict insertion order) --------
  const float* x        = (const float*)d_in[0];
  const float* enc_we_w = (const float*)d_in[1];
  const float* enc_we_b = (const float*)d_in[2];
  const float* enc_ln_g = (const float*)d_in[19];
  const float* enc_ln_b = (const float*)d_in[20];
  const float* dec_we_w = (const float*)d_in[21];
  const float* dec_we_b = (const float*)d_in[22];
  const float* dec_ln_g = (const float*)d_in[39];
  const float* dec_ln_b = (const float*)d_in[40];
  const float* dec_ol_w = (const float*)d_in[41];
  const float* dec_ol_b = (const float*)d_in[42];

  // nested layer dicts: jax pytree flatten = sorted keys (b1 first, NL*F elems);
  // detect insertion-order flatten (ln1_g first, NL*E elems) as fallback.
  bool sortedOrder = (n_in > 3) ? (in_sizes[3] == NL * F) : true;
  auto getLayer = [&](int base, int l) {
    LP p;
    auto leaf = [&](int idx, size_t per) {
      return (const float*)d_in[base + idx] + (size_t)l * per;
    };
    size_t EE = (size_t)E * E, EF = (size_t)E * F;
    if (sortedOrder) {  // b1,b2,bk,bo,bq,bv,ln1_b,ln1_g,ln2_b,ln2_g,w1,w2,wk,wo,wq,wv
      p.b1 = leaf(0, F);  p.b2 = leaf(1, E);  p.bk = leaf(2, E);  p.bo = leaf(3, E);
      p.bq = leaf(4, E);  p.bv = leaf(5, E);  p.ln1_b = leaf(6, E); p.ln1_g = leaf(7, E);
      p.ln2_b = leaf(8, E); p.ln2_g = leaf(9, E);
      p.w1 = leaf(10, EF); p.w2 = leaf(11, EF);
      p.wk = leaf(12, EE); p.wo = leaf(13, EE); p.wq = leaf(14, EE); p.wv = leaf(15, EE);
    } else {            // ln1_g,ln1_b,wq,bq,wk,bk,wv,bv,wo,bo,ln2_g,ln2_b,w1,b1,w2,b2
      p.ln1_g = leaf(0, E); p.ln1_b = leaf(1, E);
      p.wq = leaf(2, EE); p.bq = leaf(3, E);
      p.wk = leaf(4, EE); p.bk = leaf(5, E);
      p.wv = leaf(6, EE); p.bv = leaf(7, E);
      p.wo = leaf(8, EE); p.bo = leaf(9, E);
      p.ln2_g = leaf(10, E); p.ln2_b = leaf(11, E);
      p.w1 = leaf(12, EF); p.b1 = leaf(13, F);
      p.w2 = leaf(14, EF); p.b2 = leaf(15, E);
    }
    return p;
  };

  // -------- workspace carve --------
  char* ws = (char*)d_ws;
  size_t off = 0;
  auto alloc = [&](size_t bytes) {
    void* p = ws + off;
    off = (off + bytes + 255) & ~(size_t)255;
    return p;
  };
  float* carry = (float*)alloc((size_t)M * E * 4);
  h16* hbuf  = (h16*)alloc((size_t)M * E * 2);   // encoder output f16
  h16* z16   = (h16*)alloc((size_t)M * E * 2);
  h16* q16   = (h16*)alloc((size_t)M * E * 2);
  h16* k16   = (h16*)alloc((size_t)M * E * 2);
  h16* vT16  = (h16*)alloc((size_t)E * M * 2);   // V^T [E, M]
  h16* c16   = (h16*)alloc((size_t)M * E * 2);
  h16* hd16  = (h16*)alloc((size_t)M * F * 2);
  h16* x16   = (h16*)alloc((size_t)M * DI * 2);
  h16* wT    = (h16*)alloc((size_t)E * F * 2);   // reused transposed weight
  (void)ws_size; (void)out_size;

  // -------- helpers --------
  auto gemm = [&](const h16* A, const float* W, int K, int N, const float* bias,
                  const float* res, float* outF, h16* outH, h16* outHT, int relu) {
    k_transpose_h<<<dim3(N / 32, K / 32), dim3(32, 8), 0, stream>>>(W, wT, K, N);
    int tiles = (M / 16) * (N / 64);
    k_gemm<<<(tiles + 7) / 8, 256, 0, stream>>>(A, wT, bias, res, outF, outH,
                                                outHT, M, N, K, relu);
  };
  auto ln = [&](const float* in, const float* g, const float* b,
                float* oF, h16* oH) {
    k_ln<<<M, 128, 0, stream>>>(in, g, b, oF, oH, E);
  };
  auto posEmbed = [&]() {
    int tot = M * E;
    k_pos<<<(tot + 255) / 256, 256, 0, stream>>>(carry, S, E, kInvSqrtE, tot);
  };
  auto attn = [&]() {
    k_attn<<<dim3(S / 16, NH, B), 32, 0, stream>>>(q16, k16, vT16, c16, S, E, M,
                                                   kInvSqrtE);
  };
  auto layerBody = [&](const LP& p, const h16* kvsrc) {
    ln(carry, p.ln1_g, p.ln1_b, nullptr, z16);
    gemm(z16,   p.wq, E, E, p.bq, nullptr, nullptr, q16, nullptr, 0);
    gemm(kvsrc, p.wk, E, E, p.bk, nullptr, nullptr, k16, nullptr, 0);
    gemm(kvsrc, p.wv, E, E, p.bv, nullptr, nullptr, nullptr, vT16, 0);
    attn();
    gemm(c16, p.wo, E, E, p.bo, carry, carry, nullptr, nullptr, 0);
    ln(carry, p.ln2_g, p.ln2_b, nullptr, z16);
    gemm(z16,  p.w1, E, F, p.b1, nullptr, nullptr, hd16, nullptr, 1);
    gemm(hd16, p.w2, F, E, p.b2, carry, carry, nullptr, nullptr, 0);
  };

  // -------- pipeline --------
  {
    int n = M * DI;
    k_cvt_h<<<(n + 255) / 256, 256, 0, stream>>>(x, x16, n);
  }

  // Encoders
  gemm(x16, enc_we_w, DI, E, enc_we_b, nullptr, carry, nullptr, nullptr, 0);
  posEmbed();
  for (int l = 0; l < NL; ++l) layerBody(getLayer(3, l), z16);
  ln(carry, enc_ln_g, enc_ln_b, nullptr, hbuf);

  // Decoders (cross-attention reads K/V from encoder output hbuf)
  gemm(x16, dec_we_w, DI, E, dec_we_b, nullptr, carry, nullptr, nullptr, 0);
  posEmbed();
  for (int l = 0; l < NL; ++l) layerBody(getLayer(23, l), hbuf);
  ln(carry, dec_ln_g, dec_ln_b, nullptr, z16);

  // Output projection -> f32 d_out
  gemm(z16, dec_ol_w, E, DI, dec_ol_b, nullptr, (float*)d_out, nullptr, nullptr, 0);
}
